// ExpDock_1254130450947
// MI455X (gfx1250) — matrix operand order repaired and checked by reading, not compiled, not run on hardware
//
#include <hip/hip_runtime.h>

#define BS 4
#define NP 1000
#define HD 128
#define LAYERS 4
#define RBF_DIM 20
#define EDGE_D 16
#define NODE_D 64
#define K_NEI 9
#define NKP 12
#define NN (BS * 2 * NP)       /* 8000 nodes  */
#define EE (NN * K_NEI)        /* 72000 edges */
#define STDV 10.0f
#define SIGMA_C 25.0f
#define GAMMA_C 10.0f
#define PI_F 3.14159265358979323846f
#define SI_LD 132              /* Si row stride, padded from 129 for alignment */

typedef __attribute__((ext_vector_type(16))) _Float16 v16h;
typedef __attribute__((ext_vector_type(8)))  _Float16 h8;
typedef __attribute__((ext_vector_type(8)))  float    v8f;

static inline int cdiv(int a, int b) { return (a + b - 1) / b; }

__device__ __forceinline__ float siluf(float z) { return z / (1.f + expf(-z)); }

#define WMMA(A_, B_, C_) \
  __builtin_amdgcn_wmma_f32_16x16x32_f16(false, (A_), false, (B_), (short)0, (C_), false, false)

// A fragment (16x32 f16, ISA 7.12.2) from contiguous f32 run (needs 16B align).
__device__ __forceinline__ v16h load_a_f32(const float* __restrict__ p) {
  float4 r0 = *(const float4*)(p);
  float4 r1 = *(const float4*)(p + 4);
  float4 r2 = *(const float4*)(p + 16);
  float4 r3 = *(const float4*)(p + 20);
  v16h a;
  a[0] = (_Float16)r0.x;  a[1] = (_Float16)r0.y;  a[2] = (_Float16)r0.z;  a[3] = (_Float16)r0.w;
  a[4] = (_Float16)r1.x;  a[5] = (_Float16)r1.y;  a[6] = (_Float16)r1.z;  a[7] = (_Float16)r1.w;
  a[8] = (_Float16)r2.x;  a[9] = (_Float16)r2.y;  a[10] = (_Float16)r2.z; a[11] = (_Float16)r2.w;
  a[12] = (_Float16)r3.x; a[13] = (_Float16)r3.y; a[14] = (_Float16)r3.z; a[15] = (_Float16)r3.w;
  return a;
}

// A fragment from f16 data: two 16-byte loads, no converts.
__device__ __forceinline__ v16h load_a_f16(const _Float16* __restrict__ p) {
  h8 lo = *(const h8*)(p);
  h8 hi = *(const h8*)(p + 16);
  v16h a;
#pragma unroll
  for (int i = 0; i < 8; ++i) { a[i] = lo[i]; a[i + 8] = hi[i]; }
  return a;
}

// B fragments for 2 consecutive 16-wide N tiles; raw-staged for clause batching.
__device__ __forceinline__ void load_b2(const float* __restrict__ q, int ldb,
                                        v16h& b0, v16h& b1) {
  float r0[16], r1[16];
  const float* r = q;
#pragma unroll
  for (int i = 0; i < 16; ++i) { const float* rr = r + (long)((i < 8) ? i : (i + 8)) * ldb; r0[i] = rr[0]; r1[i] = rr[16]; }
#pragma unroll
  for (int i = 0; i < 16; ++i) { b0[i] = (_Float16)r0[i]; b1[i] = (_Float16)r1[i]; }
}

// ---------------------------------------------------------------------------
// gemm_f32: C[M,N] = act(A @ B + bias) (+residual). A f32, contiguous rows,
// lda % 4 == 0. M%16==0, N%32==0, K ragged (clamped tail). Dense C (ldc=N).
// Optional f16 mirror C16. One wave -> 16x32 strip (2 WMMA tiles).
// ---------------------------------------------------------------------------
__global__ void gemm_f32(const float* __restrict__ A, int lda,
                         const float* __restrict__ B,
                         const float* __restrict__ bias,
                         const float* __restrict__ residual,
                         float* __restrict__ C, _Float16* __restrict__ C16,
                         int M, int N, int K, int act) {
  int lane = threadIdx.x & 31;
  int wid = (blockIdx.x * blockDim.x + threadIdx.x) >> 5;
  int groupsN = N >> 5;
  if (wid >= (M >> 4) * groupsN) return;
  int tm = wid / groupsN, tg = wid % groupsN;
  int row0 = tm * 16, col0 = tg * 32;

  int arow = row0 + (lane & 15);
  const float* Arow = A + (long)arow * lda;
  int kbaseA = (lane >> 4) * 8;
  int ncol = lane & 15;
  int kbaseB = (lane >> 4) * 8;

  const float* Ap = Arow + kbaseA;
  const float* Bp = B + (long)kbaseB * N + col0 + ncol;

  v8f acc0 = {}, acc1 = {};
  int K32 = K & ~31;
  for (int k0 = 0; k0 < K32; k0 += 32) {
    v16h af = load_a_f32(Ap);
    v16h b0, b1;
    load_b2(Bp, N, b0, b1);
    acc0 = WMMA(af, b0, acc0);
    acc1 = WMMA(af, b1, acc1);
    Ap += 32;
    Bp += (long)32 * N;
  }
  if (K32 < K) {  // ragged tail: clamp addresses, select values
    v16h af, b0, b1;
    float ra[16], r0[16], r1[16];
#pragma unroll
    for (int i = 0; i < 16; ++i) {
      int k = K32 + ((i < 8) ? (kbaseA + i) : (16 + kbaseA + (i - 8)));
      int kk = (k < K) ? k : (K - 1);
      ra[i] = Arow[kk];
      if (!(k < K)) ra[i] = 0.f;
    }
    const float* q0 = B + col0 + ncol;
#pragma unroll
    for (int i = 0; i < 16; ++i) {
      int k = K32 + ((i < 8) ? (kbaseB + i) : (16 + kbaseB + (i - 8)));
      int kk = (k < K) ? k : (K - 1);
      const float* r = q0 + (long)kk * N;
      float m0 = r[0], m1 = r[16];
      bool ok = k < K;
      r0[i] = ok ? m0 : 0.f;
      r1[i] = ok ? m1 : 0.f;
    }
#pragma unroll
    for (int i = 0; i < 16; ++i) {
      af[i] = (_Float16)ra[i]; b0[i] = (_Float16)r0[i]; b1[i] = (_Float16)r1[i];
    }
    acc0 = WMMA(af, b0, acc0);
    acc1 = WMMA(af, b1, acc1);
  }

  int mofs = (lane >> 4) * 8;
  v8f accs[2] = {acc0, acc1};
#pragma unroll
  for (int t = 0; t < 2; ++t) {
    int cn = col0 + t * 16 + ncol;
    float bv = bias ? bias[cn] : 0.f;
    long base = (long)(row0 + mofs) * N + cn;
#pragma unroll
    for (int r = 0; r < 8; ++r) {
      float z = accs[t][r] + bv;
      if (act == 1) z = siluf(z);
      long idx = base + (long)r * N;
      if (residual) z += residual[idx];
      C[idx] = z;
      if (C16) C16[idx] = (_Float16)z;
    }
  }
}

// ---------------------------------------------------------------------------
// gemm_f16a: C[M,N] = A @ B. A already f16 (lda in elements), no bias/act.
// Used for mx = m16 @ x_W and mv = m16 @ v_W. M%16==0, N%32==0, K%32==0.
// ---------------------------------------------------------------------------
__global__ void gemm_f16a(const _Float16* __restrict__ A, int lda,
                          const float* __restrict__ B,
                          float* __restrict__ C, int M, int N, int K) {
  int lane = threadIdx.x & 31;
  int wid = (blockIdx.x * blockDim.x + threadIdx.x) >> 5;
  int groupsN = N >> 5;
  if (wid >= (M >> 4) * groupsN) return;
  int tm = wid / groupsN, tg = wid % groupsN;
  int row0 = tm * 16, col0 = tg * 32;

  int kbaseA = (lane >> 4) * 8;
  int ncol = lane & 15;
  int kbaseB = (lane >> 4) * 8;
  const _Float16* Ap = A + (long)(row0 + (lane & 15)) * lda + kbaseA;
  const float* Bp = B + (long)kbaseB * N + col0 + ncol;

  v8f acc0 = {}, acc1 = {};
  for (int k0 = 0; k0 < K; k0 += 32) {
    v16h af = load_a_f16(Ap);
    v16h b0, b1;
    load_b2(Bp, N, b0, b1);
    acc0 = WMMA(af, b0, acc0);
    acc1 = WMMA(af, b1, acc1);
    Ap += 32;
    Bp += (long)32 * N;
  }
  int mofs = (lane >> 4) * 8;
  v8f accs[2] = {acc0, acc1};
#pragma unroll
  for (int t = 0; t < 2; ++t) {
    int cn = col0 + t * 16 + ncol;
    long base = (long)(row0 + mofs) * N + cn;
#pragma unroll
    for (int r = 0; r < 8; ++r) C[base + (long)r * N] = accs[t][r];
  }
}

// ---------------------------------------------------------------------------
// gemm_v3: equivariant-vector contraction, hardcoded strides.
// C[m,k',c] over one component c: A element (m,k) at A[m*384 + k*3],
// B 128x128 dense, C element at C[m*384 + n*3]. M=NN, N=K=128.
// A loads: one base per chunk + constant immediate offsets.
// ---------------------------------------------------------------------------
__global__ void gemm_v3(const float* __restrict__ A, const float* __restrict__ B,
                        float* __restrict__ C) {
  int lane = threadIdx.x & 31;
  int wid = (blockIdx.x * blockDim.x + threadIdx.x) >> 5;
  if (wid >= (NN / 16) * 4) return;
  int tm = wid >> 2, tg = wid & 3;
  int row0 = tm * 16, col0 = tg * 32;

  int kbaseA = (lane >> 4) * 8;
  int ncol = lane & 15;
  int kbaseB = (lane >> 4) * 8;
  const float* Ap = A + (long)(row0 + (lane & 15)) * 384 + kbaseA * 3;
  const float* Bp = B + (long)kbaseB * HD + col0 + ncol;

  v8f acc0 = {}, acc1 = {};
#pragma unroll
  for (int k0 = 0; k0 < HD; k0 += 32) {
    float ra[16];
#pragma unroll
    for (int i = 0; i < 16; ++i) {
      int off = (i < 8) ? i : (8 + i);   /* k-offset within chunk */
      ra[i] = Ap[off * 3];
    }
    v16h af, b0, b1;
#pragma unroll
    for (int i = 0; i < 16; ++i) af[i] = (_Float16)ra[i];
    load_b2(Bp, HD, b0, b1);
    acc0 = WMMA(af, b0, acc0);
    acc1 = WMMA(af, b1, acc1);
    Ap += 96;
    Bp += 32 * HD;
  }
  int mofs = (lane >> 4) * 8;
  v8f accs[2] = {acc0, acc1};
#pragma unroll
  for (int t = 0; t < 2; ++t) {
    int cn = col0 + t * 16 + ncol;
    long base = (long)(row0 + mofs) * 384 + (long)cn * 3;
#pragma unroll
    for (int r = 0; r < 8; ++r) C[base + (long)r * 384] = accs[t][r];
  }
}

// ---------------------------------------------------------------------------
// msg_gemm: m16 = f16(silu(concat[h[src], h[dst], rbf, edge_attr] @ W + b))
// h gathered from f16 mirror (two 16B loads per chunk, no cvt).
// ---------------------------------------------------------------------------
__global__ void msg_gemm(const _Float16* __restrict__ h16,
                         const float* __restrict__ rbf,
                         const float* __restrict__ eattr,
                         const int* __restrict__ src,
                         const int* __restrict__ dst,
                         const float* __restrict__ W,   /* 292 x 128 */
                         const float* __restrict__ bias,/* 128 */
                         _Float16* __restrict__ m16) {
  const int Kdim = 2 * HD + RBF_DIM + EDGE_D; /* 292 */
  int lane = threadIdx.x & 31;
  int wid = (blockIdx.x * blockDim.x + threadIdx.x) >> 5;
  if (wid >= (EE / 16) * 4) return;
  int tm = wid >> 2, tg = wid & 3;
  int row0 = tm * 16, col0 = tg * 32;

  int e = row0 + (lane & 15);
  int se = src[e], de = dst[e];
  const _Float16* hs = h16 + (long)se * HD;
  const _Float16* hd = h16 + (long)de * HD;
  const float* rb = rbf + (long)e * RBF_DIM;
  const float* ea = eattr + (long)e * EDGE_D;
  int kbaseA = (lane >> 4) * 8;
  int ncol = lane & 15;
  int kbaseB = (lane >> 4) * 8;

  const float* Bp = W + (long)kbaseB * HD + col0 + ncol;
  v8f acc0 = {}, acc1 = {};
  v16h af, b0, b1;

#pragma unroll
  for (int c = 0; c < 4; ++c) {  // k = 0..127 : h[src]
    af = load_a_f16(hs + c * 32 + kbaseA);
    load_b2(Bp, HD, b0, b1);
    Bp += 32 * HD;
    acc0 = WMMA(af, b0, acc0);
    acc1 = WMMA(af, b1, acc1);
  }
#pragma unroll
  for (int c = 0; c < 4; ++c) {  // k = 128..255 : h[dst]
    af = load_a_f16(hd + c * 32 + kbaseA);
    load_b2(Bp, HD, b0, b1);
    Bp += 32 * HD;
    acc0 = WMMA(af, b0, acc0);
    acc1 = WMMA(af, b1, acc1);
  }
  {  // k = 256..287 : rbf (256..275) / edge_attr (276..287), all in-bounds
    float ra[16];
#pragma unroll
    for (int i = 0; i < 16; ++i) {
      int k = 256 + ((i < 8) ? (kbaseA + i) : (16 + kbaseA + (i - 8)));
      const float* p = (k < 276) ? (rb + (k - 256)) : (ea + (k - 276));
      ra[i] = *p;
    }
#pragma unroll
    for (int i = 0; i < 16; ++i) af[i] = (_Float16)ra[i];
    load_b2(Bp, HD, b0, b1);
    Bp += 32 * HD;
    acc0 = WMMA(af, b0, acc0);
    acc1 = WMMA(af, b1, acc1);
  }
  {  // k = 288..319 : edge_attr tail (valid k < 292), clamp + select
    float ra[16], r0[16], r1[16];
#pragma unroll
    for (int i = 0; i < 16; ++i) {
      int k = 288 + ((i < 8) ? (kbaseA + i) : (16 + kbaseA + (i - 8)));
      int idx = k - 276;
      idx = (idx > 15) ? 15 : idx;
      float v = ea[idx];
      ra[i] = (k < Kdim) ? v : 0.f;
    }
    const float* q0 = W + col0 + ncol;
#pragma unroll
    for (int i = 0; i < 16; ++i) {
      int k = 288 + ((i < 8) ? (kbaseB + i) : (16 + kbaseB + (i - 8)));
      int kk = (k < Kdim) ? k : (Kdim - 1);
      const float* r = q0 + (long)kk * HD;
      float m0 = r[0], m1 = r[16];
      bool ok = k < Kdim;
      r0[i] = ok ? m0 : 0.f;
      r1[i] = ok ? m1 : 0.f;
    }
#pragma unroll
    for (int i = 0; i < 16; ++i) {
      af[i] = (_Float16)ra[i]; b0[i] = (_Float16)r0[i]; b1[i] = (_Float16)r1[i];
    }
    acc0 = WMMA(af, b0, acc0);
    acc1 = WMMA(af, b1, acc1);
  }

  int mofs = (lane >> 4) * 8;
  v8f accs[2] = {acc0, acc1};
#pragma unroll
  for (int t = 0; t < 2; ++t) {
    int cn = col0 + t * 16 + ncol;
    float bv = bias[cn];
    long base = (long)(row0 + mofs) * HD + cn;
#pragma unroll
    for (int r = 0; r < 8; ++r)
      m16[base + (long)r * HD] = (_Float16)siluf(accs[t][r] + bv);
  }
}

// ---------------------------------------------------------------------------
// Elementwise / gather / reduction kernels
// ---------------------------------------------------------------------------
__global__ void prep_nodes(const float* __restrict__ X, const float* __restrict__ center,
                           const float* __restrict__ rot, const float* __restrict__ trans,
                           float* __restrict__ Xca, float* __restrict__ ori) {
  int n = blockIdx.x * blockDim.x + threadIdx.x;
  if (n >= NN) return;
  int b = n / (2 * NP);
  int side = (n / NP) & 1;
  const float* c = center + b * 3;
  float p0 = X[(long)n * 12 + 3] - c[0];
  float p1 = X[(long)n * 12 + 4] - c[1];
  float p2 = X[(long)n * 12 + 5] - c[2];
  ori[n * 3 + 0] = p0 / STDV; ori[n * 3 + 1] = p1 / STDV; ori[n * 3 + 2] = p2 / STDV;
  float q0 = p0, q1 = p1, q2 = p2;
  if (side == 0) {
    const float* R = rot + b * 9;
    const float* t = trans + b * 3;
    q0 = p0 * R[0] + p1 * R[3] + p2 * R[6] + STDV * t[0];
    q1 = p0 * R[1] + p1 * R[4] + p2 * R[7] + STDV * t[1];
    q2 = p0 * R[2] + p1 * R[5] + p2 * R[8] + STDV * t[2];
  }
  Xca[n * 3 + 0] = q0 / STDV; Xca[n * 3 + 1] = q1 / STDV; Xca[n * 3 + 2] = q2 / STDV;
}

__global__ void prep_kp(const float* __restrict__ keypoints, const float* __restrict__ center,
                        const float* __restrict__ rot, const float* __restrict__ trans,
                        float* __restrict__ kp, float* __restrict__ tkp) {
  int i = blockIdx.x * blockDim.x + threadIdx.x;
  if (i >= BS * NKP) return;
  int b = i / NKP;
  const float* c = center + b * 3;
  float p0 = keypoints[i * 3 + 0] - c[0];
  float p1 = keypoints[i * 3 + 1] - c[1];
  float p2 = keypoints[i * 3 + 2] - c[2];
  kp[i * 3 + 0] = p0 / STDV; kp[i * 3 + 1] = p1 / STDV; kp[i * 3 + 2] = p2 / STDV;
  const float* R = rot + b * 9;
  const float* t = trans + b * 3;
  tkp[i * 3 + 0] = (p0 * R[0] + p1 * R[3] + p2 * R[6] + STDV * t[0]) / STDV;
  tkp[i * 3 + 1] = (p0 * R[1] + p1 * R[4] + p2 * R[7] + STDV * t[1]) / STDV;
  tkp[i * 3 + 2] = (p0 * R[2] + p1 * R[5] + p2 * R[8] + STDV * t[2]) / STDV;
}

__global__ void edge_geom(const float* __restrict__ Xca, const int* __restrict__ src,
                          const int* __restrict__ dst, float* __restrict__ dirv,
                          float* __restrict__ rbf) {
  int e = blockIdx.x * blockDim.x + threadIdx.x;
  if (e >= EE) return;
  int s = src[e], d = dst[e];
  float dx = Xca[d * 3 + 0] - Xca[s * 3 + 0];
  float dy = Xca[d * 3 + 1] - Xca[s * 3 + 1];
  float dz = Xca[d * 3 + 2] - Xca[s * 3 + 2];
  float nrm = sqrtf(dx * dx + dy * dy + dz * dz) + 1e-8f;
  dirv[e * 3 + 0] = dx / nrm; dirv[e * 3 + 1] = dy / nrm; dirv[e * 3 + 2] = dz / nrm;
  float inv = 1.f / nrm;
#pragma unroll
  for (int j = 0; j < RBF_DIM; ++j)
    rbf[(long)e * RBF_DIM + j] = sinf(nrm * (float)(j + 1) * PI_F) * inv;
}

__global__ void agg_kernel(const _Float16* __restrict__ m16, float* __restrict__ agg) {
  int idx = blockIdx.x * blockDim.x + threadIdx.x;
  if (idx >= NN * HD) return;
  int n = idx >> 7, hc = idx & 127;
  const _Float16* base = m16 + (long)n * K_NEI * HD + hc;
  float s = 0.f;
#pragma unroll
  for (int j = 0; j < K_NEI; ++j) s += (float)base[j * HD];
  agg[idx] = s;
}

__global__ void concat2(const float* __restrict__ a, const float* __restrict__ b,
                        float* __restrict__ out, int n, int da, int db) {
  int i = blockIdx.x * blockDim.x + threadIdx.x;
  int dt = da + db;
  if (i >= n * dt) return;
  int row = i / dt, col = i - row * dt;
  out[i] = (col < da) ? a[(long)row * da + col] : b[(long)row * db + (col - da)];
}

__global__ void vupd_kernel(const float* __restrict__ v, const float* __restrict__ mx,
                            const float* __restrict__ mv, const float* __restrict__ dirv,
                            const int* __restrict__ src, float* __restrict__ vout) {
  int idx = blockIdx.x * blockDim.x + threadIdx.x;
  if (idx >= NN * HD) return;
  int n = idx >> 7, hc = idx & 127;
  float a0 = v[(long)n * 384 + hc * 3 + 0];
  float a1 = v[(long)n * 384 + hc * 3 + 1];
  float a2 = v[(long)n * 384 + hc * 3 + 2];
#pragma unroll
  for (int j = 0; j < K_NEI; ++j) {
    int e = n * K_NEI + j;
    int s = src[e];
    float xm = mx[(long)e * HD + hc];
    float vm = mv[(long)e * HD + hc];
    a0 += dirv[e * 3 + 0] * xm + v[(long)s * 384 + hc * 3 + 0] * vm;
    a1 += dirv[e * 3 + 1] * xm + v[(long)s * 384 + hc * 3 + 1] * vm;
    a2 += dirv[e * 3 + 2] * xm + v[(long)s * 384 + hc * 3 + 2] * vm;
  }
  vout[(long)n * 384 + hc * 3 + 0] = a0;
  vout[(long)n * 384 + hc * 3 + 1] = a1;
  vout[(long)n * 384 + hc * 3 + 2] = a2;
}

__global__ void group_mean(const float* __restrict__ x, float* __restrict__ out) {
  int g = blockIdx.x, k = threadIdx.x;
  const float* base = x + (long)g * NP * HD + k;
  float s = 0.f;
  for (int i = 0; i < NP; ++i) s += base[(long)i * HD];
  out[g * HD + k] = s * (1.f / NP);
}

__global__ void att_u(const float* __restrict__ attW, const float* __restrict__ hbar,
                      float* __restrict__ u) {
  int g = blockIdx.x, hr = threadIdx.x;
  const float* hb = hbar + (g ^ 1) * HD;
  float s = 0.f;
  for (int k = 0; k < HD; ++k) s += attW[hr * HD + k] * hb[k];
  u[g * HD + hr] = s;
}

__global__ void att_gate(const float* __restrict__ intra, const float* __restrict__ u,
                         float* __restrict__ gate) {
  int row = blockIdx.x * blockDim.x + threadIdx.x;
  if (row >= NN) return;
  int g = row / NP;
  const float* hr = intra + (long)row * HD;
  const float* ug = u + g * HD;
  float s = 0.f;
  for (int k = 0; k < HD; ++k) s += hr[k] * ug[k];
  gate[row] = 1.f / (1.f + expf(-s));
}

__global__ void h_update(const float* __restrict__ intra, const float* __restrict__ t2,
                         const float* __restrict__ gate, float* __restrict__ hnew,
                         _Float16* __restrict__ h16) {
  int idx = blockIdx.x * blockDim.x + threadIdx.x;
  if (idx >= NN * HD) return;
  int n = idx >> 7;
  float z = intra[idx] + gate[n] * t2[idx];
  hnew[idx] = z;
  h16[idx] = (_Float16)z;
}

__global__ void vnorm_kernel(const float* __restrict__ v2, float* __restrict__ out) {
  int idx = blockIdx.x * blockDim.x + threadIdx.x;
  if (idx >= NN * HD) return;
  int n = idx >> 7, k = idx & 127;
  float a = v2[(long)n * 384 + k * 3 + 0];
  float b = v2[(long)n * 384 + k * 3 + 1];
  float c = v2[(long)n * 384 + k * 3 + 2];
  out[idx] = sqrtf(a * a + b * b + c * c);
}

__global__ void split_scale(const float* __restrict__ sbuf, const float* __restrict__ v1,
                            float* __restrict__ hg, float* __restrict__ vg) {
  int idx = blockIdx.x * blockDim.x + threadIdx.x;
  if (idx >= NN * HD) return;
  int n = idx >> 7, k = idx & 127;
  hg[idx] = sbuf[(long)n * 256 + k];
  float sc = sbuf[(long)n * 256 + HD + k];
  vg[(long)n * 384 + k * 3 + 0] = v1[(long)n * 384 + k * 3 + 0] * sc;
  vg[(long)n * 384 + k * 3 + 1] = v1[(long)n * 384 + k * 3 + 1] * sc;
  vg[(long)n * 384 + k * 3 + 2] = v1[(long)n * 384 + k * 3 + 2] * sc;
}

__global__ void cmean_kernel(const float* __restrict__ Xca, float* __restrict__ cmean) {
  int i = threadIdx.x;
  if (i >= 24) return;
  int g = i / 3, d = i % 3;
  float s = 0.f;
  for (int j = 0; j < NP; ++j) s += Xca[((long)g * NP + j) * 3 + d];
  cmean[i] = s * (1.f / NP);
}

__global__ void tred_kernel(const float* __restrict__ vg, const float* __restrict__ hg,
                            float* __restrict__ T) {
  int idx = blockIdx.x * blockDim.x + threadIdx.x;
  if (idx >= 8 * HD) return;
  int g = idx >> 7, hc = idx & 127;
  float s0 = 0.f, s1 = 0.f, s2 = 0.f;
  for (int j = 0; j < NP; ++j) {
    long r = (long)g * NP + j;
    float hh = hg[r * HD + hc];
    s0 += vg[r * 384 + hc * 3 + 0] * hh;
    s1 += vg[r * 384 + hc * 3 + 1] * hh;
    s2 += vg[r * 384 + hc * 3 + 2] * hh;
  }
  T[g * 384 + hc * 3 + 0] = s0;
  T[g * 384 + hc * 3 + 1] = s1;
  T[g * 384 + hc * 3 + 2] = s2;
}

__global__ void ykp_kernel(const float* __restrict__ finalW, const float* __restrict__ T,
                           float* __restrict__ Yk) {
  int t = threadIdx.x;
  if (t >= 8 * NKP) return;
  int g = t / NKP, k = t % NKP;
#pragma unroll
  for (int c = 0; c < 3; ++c) {
    float s = 0.f;
    for (int h = 0; h < HD; ++h) s += finalW[h * NKP + k] * T[g * 384 + h * 3 + c];
    Yk[g * 36 + k * 3 + c] = s;
  }
}

__global__ void si_kernel(const float* __restrict__ vg, const float* __restrict__ Xca,
                          const float* __restrict__ cmean, float* __restrict__ Si) {
  int n = blockIdx.x * blockDim.x + threadIdx.x;
  if (n >= NN) return;
  int g = n / NP;
  for (int h = 0; h < HD; ++h) {
    float a = vg[(long)n * 384 + h * 3 + 0];
    float b = vg[(long)n * 384 + h * 3 + 1];
    float c = vg[(long)n * 384 + h * 3 + 2];
    Si[(long)n * SI_LD + h] = sqrtf(a * a + b * b + c * c);
  }
  float dx = Xca[n * 3 + 0] - cmean[g * 3 + 0];
  float dy = Xca[n * 3 + 1] - cmean[g * 3 + 1];
  float dz = Xca[n * 3 + 2] - cmean[g * 3 + 2];
  Si[(long)n * SI_LD + 128] = sqrtf(dx * dx + dy * dy + dz * dz);
  Si[(long)n * SI_LD + 129] = 0.f;
  Si[(long)n * SI_LD + 130] = 0.f;
  Si[(long)n * SI_LD + 131] = 0.f;
}

__global__ void denomrow_kernel(const float* __restrict__ d1, const float* __restrict__ w2,
                                const float* __restrict__ b2, float* __restrict__ dr) {
  int n = blockIdx.x * blockDim.x + threadIdx.x;
  if (n >= NN) return;
  float s = b2[0];
  for (int j = 0; j < 64; ++j) s += d1[(long)n * 64 + j] * w2[j];
  dr[n] = s;
}

__global__ void denom_kernel(const float* __restrict__ dr, float* __restrict__ denom) {
  int g = blockIdx.x;
  if (threadIdx.x) return;
  float s = 0.f;
  for (int i = 0; i < NP; ++i) s += dr[g * NP + i];
  denom[g] = s * (1.f / NP);
}

__global__ void yfinal_kernel(const float* __restrict__ Yk, const float* __restrict__ denom,
                              const float* __restrict__ cmean, float* __restrict__ Yf) {
  int t = threadIdx.x;
  if (t >= 8 * NKP) return;
  int g = t / NKP, k = t % NKP;
#pragma unroll
  for (int c = 0; c < 3; ++c)
    Yf[g * 36 + k * 3 + c] = Yk[g * 36 + k * 3 + c] / denom[g] + cmean[g * 3 + c];
}

// ---------------------------------------------------------------------------
// Kabsch (3x3 SVD via Jacobi), per-batch losses
// ---------------------------------------------------------------------------
__device__ void jacobi3(float A[3][3], float V[3][3]) {
  for (int i = 0; i < 3; ++i)
    for (int j = 0; j < 3; ++j) V[i][j] = (i == j) ? 1.f : 0.f;
  for (int sweep = 0; sweep < 12; ++sweep) {
    for (int p = 0; p < 2; ++p)
      for (int q = p + 1; q < 3; ++q) {
        float apq = A[p][q];
        if (fabsf(apq) < 1e-20f) continue;
        float tau = (A[q][q] - A[p][p]) / (2.f * apq);
        float t = ((tau >= 0.f) ? 1.f : -1.f) / (fabsf(tau) + sqrtf(1.f + tau * tau));
        float c = 1.f / sqrtf(1.f + t * t), s = t * c;
        for (int k = 0; k < 3; ++k) {
          float akp = A[k][p], akq = A[k][q];
          A[k][p] = c * akp - s * akq;
          A[k][q] = s * akp + c * akq;
        }
        for (int k = 0; k < 3; ++k) {
          float apk = A[p][k], aqk = A[q][k];
          A[p][k] = c * apk - s * aqk;
          A[q][k] = s * apk + c * aqk;
        }
        for (int k = 0; k < 3; ++k) {
          float vkp = V[k][p], vkq = V[k][q];
          V[k][p] = c * vkp - s * vkq;
          V[k][q] = s * vkp + c * vkq;
        }
      }
  }
}

__device__ float det3f(const float M[3][3]) {
  return M[0][0] * (M[1][1] * M[2][2] - M[1][2] * M[2][1])
       - M[0][1] * (M[1][0] * M[2][2] - M[1][2] * M[2][0])
       + M[0][2] * (M[1][0] * M[2][1] - M[1][1] * M[2][0]);
}

__global__ void kabsch_kernel(const float* __restrict__ Yf, const float* __restrict__ tkp,
                              const float* __restrict__ kp, const float* __restrict__ rot,
                              const float* __restrict__ trans, float* __restrict__ RT,
                              float* __restrict__ otdock) {
  int b = blockIdx.x;
  if (threadIdx.x) return;
  const float* Y1 = Yf + (b * 2 + 0) * 36;
  const float* Y2 = Yf + (b * 2 + 1) * 36;
  const float* P1 = tkp + b * 36;
  const float* P2 = kp + b * 36;
  float ot = 0.f;
  for (int n = 0; n < NKP; ++n) {
    float best = 1e30f;
    for (int m2 = 0; m2 < NKP; ++m2) {
      float d0 = Y1[n * 3] - P1[m2 * 3], d1 = Y1[n * 3 + 1] - P1[m2 * 3 + 1],
            d2 = Y1[n * 3 + 2] - P1[m2 * 3 + 2];
      float d = d0 * d0 + d1 * d1 + d2 * d2;
      best = fminf(best, d);
    }
    ot += best;
  }
  for (int n = 0; n < NKP; ++n) {
    float best = 1e30f;
    for (int m2 = 0; m2 < NKP; ++m2) {
      float d0 = Y2[n * 3] - P2[m2 * 3], d1 = Y2[n * 3 + 1] - P2[m2 * 3 + 1],
            d2 = Y2[n * 3 + 2] - P2[m2 * 3 + 2];
      float d = d0 * d0 + d1 * d1 + d2 * d2;
      best = fminf(best, d);
    }
    ot += best;
  }
  ot /= (float)(NKP * 3);

  float c1[3] = {0, 0, 0}, c2[3] = {0, 0, 0};
  for (int n = 0; n < NKP; ++n)
    for (int d = 0; d < 3; ++d) { c1[d] += Y1[n * 3 + d]; c2[d] += Y2[n * 3 + d]; }
  for (int d = 0; d < 3; ++d) { c1[d] /= NKP; c2[d] /= NKP; }
  float Hm[3][3] = {{0, 0, 0}, {0, 0, 0}, {0, 0, 0}};
  for (int n = 0; n < NKP; ++n)
    for (int i = 0; i < 3; ++i)
      for (int j = 0; j < 3; ++j)
        Hm[i][j] += (Y1[n * 3 + i] - c1[i]) * (Y2[n * 3 + j] - c2[j]);
  float S[3][3], V[3][3];
  for (int i = 0; i < 3; ++i)
    for (int j = 0; j < 3; ++j) {
      float s = 0.f;
      for (int k = 0; k < 3; ++k) s += Hm[k][i] * Hm[k][j];
      S[i][j] = s;
    }
  jacobi3(S, V);
  float w[3] = {S[0][0], S[1][1], S[2][2]};
  for (int i = 0; i < 2; ++i)
    for (int j = 0; j < 2 - i; ++j)
      if (w[j] < w[j + 1]) {
        float tw = w[j]; w[j] = w[j + 1]; w[j + 1] = tw;
        for (int k = 0; k < 3; ++k) { float tv = V[k][j]; V[k][j] = V[k][j + 1]; V[k][j + 1] = tv; }
      }
  float U[3][3];
  for (int k = 0; k < 3; ++k) {
    float sv = sqrtf(fmaxf(w[k], 0.f));
    float inv = 1.f / fmaxf(sv, 1e-12f);
    for (int i = 0; i < 3; ++i) {
      float s = 0.f;
      for (int j = 0; j < 3; ++j) s += Hm[i][j] * V[j][k];
      U[i][k] = s * inv;
    }
  }
  float dsign = (det3f(Hm) >= 0.f) ? 1.f : -1.f;
  float f[3] = {1.f, 1.f, dsign};
  float R[3][3];
  for (int i = 0; i < 3; ++i)
    for (int j = 0; j < 3; ++j) {
      float s = 0.f;
      for (int k = 0; k < 3; ++k) s += f[k] * U[i][k] * V[j][k];
      R[i][j] = s;
    }
  float t[3];
  for (int j = 0; j < 3; ++j)
    t[j] = c2[j] - (c1[0] * R[0][j] + c1[1] * R[1][j] + c1[2] * R[2][j]);

  const float* Ri = rot + b * 9;
  const float* ti = trans + b * 3;
  float dock = 0.f;
  for (int i = 0; i < 3; ++i)
    for (int j = 0; j < 3; ++j) {
      float e = R[i][j] - Ri[j * 3 + i];
      dock += e * e;
    }
  dock /= 9.f;
  float td = 0.f;
  for (int j = 0; j < 3; ++j) {
    float tgt = -(ti[0] * Ri[j * 3 + 0] + ti[1] * Ri[j * 3 + 1] + ti[2] * Ri[j * 3 + 2]);
    float e = t[j] - tgt;
    td += e * e;
  }
  dock += td / 3.f;

  for (int i = 0; i < 3; ++i)
    for (int j = 0; j < 3; ++j) RT[b * 12 + i * 3 + j] = R[i][j];
  for (int j = 0; j < 3; ++j) RT[b * 12 + 9 + j] = t[j];
  otdock[b] = ot;
  otdock[4 + b] = dock;
}

__global__ void x1a_kernel(const float* __restrict__ Xca, const float* __restrict__ ori,
                           const float* __restrict__ RT, float* __restrict__ X1a,
                           float* __restrict__ rmsdrow) {
  int idx = blockIdx.x * blockDim.x + threadIdx.x;
  if (idx >= BS * NP) return;
  int b = idx / NP, i = idx % NP;
  const float* R = RT + b * 12;
  const float* t = R + 9;
  const float* xr = Xca + (((long)b * 2) * NP + i) * 3;
  const float* o = ori + (((long)b * 2) * NP + i) * 3;
  float rs = 0.f;
#pragma unroll
  for (int d = 0; d < 3; ++d) {
    float v = xr[0] * R[0 * 3 + d] + xr[1] * R[1 * 3 + d] + xr[2] * R[2 * 3 + d] + t[d];
    X1a[(long)idx * 3 + d] = v;
    float e = v - o[d];
    rs += e * e;
  }
  rmsdrow[idx] = rs;
}

__global__ void stab_kernel(const float* __restrict__ X1a, const float* __restrict__ Xca,
                            float* __restrict__ stabrow) {
  int idx = blockIdx.x * blockDim.x + threadIdx.x;
  if (idx >= 2 * BS * NP) return;
  int dir = idx / (BS * NP);
  int rem = idx % (BS * NP);
  int b = rem / NP, i = rem % NP;
  const float* A;
  const float* Bb;
  if (dir == 0) {
    A = X1a + ((long)b * NP + i) * 3;
    Bb = Xca + ((long)(b * 2 + 1) * NP) * 3;
  } else {
    A = Xca + ((long)(b * 2 + 1) * NP + i) * 3;
    Bb = X1a + ((long)b * NP) * 3;
  }
  float a0 = A[0] * STDV, a1 = A[1] * STDV, a2 = A[2] * STDV;
  float mmax = -3.4e38f, ssum = 0.f;
  for (int m2 = 0; m2 < NP; ++m2) {
    float d0 = a0 - Bb[m2 * 3 + 0] * STDV;
    float d1 = a1 - Bb[m2 * 3 + 1] * STDV;
    float d2 = a2 - Bb[m2 * 3 + 2] * STDV;
    float val = -(d0 * d0 + d1 * d1 + d2 * d2) / SIGMA_C;
    if (val > mmax) {
      ssum = ssum * expf(mmax - val) + 1.f;
      mmax = val;
    } else {
      ssum += expf(val - mmax);
    }
  }
  float lse = mmax + logf(ssum);
  float v = GAMMA_C + SIGMA_C * lse;
  stabrow[idx] = fmaxf(v, 0.f);
}

__global__ void final_kernel(const float* __restrict__ otdock, const float* __restrict__ rmsdrow,
                             const float* __restrict__ stabrow, float* __restrict__ out) {
  if (threadIdx.x || blockIdx.x) return;
  float ot = 0.f, dock = 0.f;
  for (int b = 0; b < BS; ++b) { ot += otdock[b]; dock += otdock[4 + b]; }
  ot /= BS; dock /= BS;
  float rmsd = 0.f;
  for (int b = 0; b < BS; ++b) {
    float s = 0.f;
    for (int i = 0; i < NP; ++i) s += rmsdrow[b * NP + i];
    rmsd += s / (float)(NP * 3);
  }
  rmsd /= BS;
  float stab = 0.f;
  for (int b = 0; b < BS; ++b) {
    float s0 = 0.f, s1 = 0.f;
    for (int i = 0; i < NP; ++i) {
      s0 += stabrow[b * NP + i];
      s1 += stabrow[BS * NP + b * NP + i];
    }
    stab += (s0 / NP + s1 / NP) / STDV;
  }
  stab /= BS;
  out[0] = ot + dock + stab;
  out[1] = ot;
  out[2] = dock;
  out[3] = stab;
  out[4] = rmsd;
}

// ---------------------------------------------------------------------------
// Host driver
// ---------------------------------------------------------------------------
static void gemmF(hipStream_t st, const float* A, int lda, const float* B,
                  const float* bias, const float* resid, float* C, _Float16* C16,
                  int M, int N, int K, int act) {
  int tiles = (M / 16) * (N / 32);
  gemm_f32<<<cdiv(tiles, 4), 128, 0, st>>>(A, lda, B, bias, resid, C, C16, M, N, K, act);
}

static void gemmH(hipStream_t st, const _Float16* A, int lda, const float* B,
                  float* C, int M, int N, int K) {
  int tiles = (M / 16) * (N / 32);
  gemm_f16a<<<cdiv(tiles, 4), 128, 0, st>>>(A, lda, B, C, M, N, K);
}

extern "C" void kernel_launch(void* const* d_in, const int* in_sizes, int n_in,
                              void* d_out, int out_size, void* d_ws, size_t ws_size,
                              hipStream_t stream) {
  (void)in_sizes; (void)n_in; (void)out_size; (void)ws_size;
  const float* X         = (const float*)d_in[0];
  const float* node_attr = (const float*)d_in[1];
  const float* edge_attr = (const float*)d_in[2];
  const float* center    = (const float*)d_in[3];
  const float* keypoints = (const float*)d_in[4];
  const float* rot       = (const float*)d_in[5];
  const float* trans     = (const float*)d_in[6];
  const float* W_in1     = (const float*)d_in[7];
  const float* b_in1     = (const float*)d_in[8];
  const float* W_in2     = (const float*)d_in[9];
  const float* b_in2     = (const float*)d_in[10];
  const float* msg_W     = (const float*)d_in[11];
  const float* msg_b     = (const float*)d_in[12];
  const float* upd_W1    = (const float*)d_in[13];
  const float* upd_b1    = (const float*)d_in[14];
  const float* upd_W2    = (const float*)d_in[15];
  const float* upd_b2    = (const float*)d_in[16];
  const float* x_W       = (const float*)d_in[17];
  const float* v_W       = (const float*)d_in[18];
  const float* att_W     = (const float*)d_in[19];
  const float* act_W1    = (const float*)d_in[20];
  const float* act_b1    = (const float*)d_in[21];
  const float* act_W2    = (const float*)d_in[22];
  const float* act_b2    = (const float*)d_in[23];
  const float* geb_Wv1   = (const float*)d_in[24];
  const float* geb_Wv2   = (const float*)d_in[25];
  const float* geb_W1    = (const float*)d_in[26];
  const float* geb_b1    = (const float*)d_in[27];
  const float* geb_W2    = (const float*)d_in[28];
  const float* geb_b2    = (const float*)d_in[29];
  const float* scale_W1  = (const float*)d_in[30];
  const float* scale_b1  = (const float*)d_in[31];
  const float* scale_W2  = (const float*)d_in[32];
  const float* scale_b2  = (const float*)d_in[33];
  const float* final_W   = (const float*)d_in[34];
  const int*   srcI      = (const int*)d_in[35];
  const int*   dstI      = (const int*)d_in[36];
  float* out = (float*)d_out;

  float* wsf = (float*)d_ws;
  size_t off = 0;
  auto alloc = [&](size_t n) { float* p = wsf + off; off += (n + 63) & ~(size_t)63; return p; };

  float* Xca   = alloc((size_t)NN * 3);
  float* ori   = alloc((size_t)NN * 3);
  float* kp    = alloc(BS * NKP * 3);
  float* tkp   = alloc(BS * NKP * 3);
  float* dirv  = alloc((size_t)EE * 3);
  float* rbf   = alloc((size_t)EE * RBF_DIM);
  float* ha    = alloc((size_t)NN * HD);
  float* hb    = alloc((size_t)NN * HD);
  float* agg   = alloc((size_t)NN * HD);
  float* intra = alloc((size_t)NN * HD);
  float* t1    = alloc((size_t)NN * HD);
  float* t2    = alloc((size_t)NN * HD);
  float* cat   = alloc((size_t)NN * 256);
  float* sbuf  = alloc((size_t)NN * 256);
  float* gate  = alloc(NN);
  float* hbar  = alloc(8 * HD);
  float* uvec  = alloc(8 * HD);
  float* va    = alloc((size_t)NN * HD * 3);
  float* vb    = alloc((size_t)NN * HD * 3);
  float* mx    = alloc((size_t)EE * HD);
  float* mv    = alloc((size_t)EE * HD);
  float* v1    = alloc((size_t)NN * HD * 3);
  float* v2    = alloc((size_t)NN * HD * 3);
  float* Si    = alloc((size_t)NN * SI_LD);
  _Float16* h16 = (_Float16*)alloc((size_t)NN * HD / 2);
  _Float16* m16 = (_Float16*)alloc((size_t)EE * HD / 2);
  float* dnr   = alloc(NN);
  float* denom = alloc(8);
  float* cmean = alloc(24);
  float* Tred  = alloc(8 * HD * 3);
  float* Yk    = alloc(8 * NKP * 3);
  float* Yf    = alloc(8 * NKP * 3);
  float* RT    = alloc(BS * 12);
  float* otdk  = alloc(8);
  float* X1a   = alloc((size_t)BS * NP * 3);
  float* rmsdr = alloc(BS * NP);
  float* stabr = alloc(2 * BS * NP);

  // ---- preprocessing ----
  prep_nodes<<<cdiv(NN, 256), 256, 0, stream>>>(X, center, rot, trans, Xca, ori);
  prep_kp<<<1, 64, 0, stream>>>(keypoints, center, rot, trans, kp, tkp);
  edge_geom<<<cdiv(EE, 256), 256, 0, stream>>>(Xca, srcI, dstI, dirv, rbf);
  hipMemsetAsync(va, 0, (size_t)NN * HD * 3 * sizeof(float), stream);

  // ---- h = silu(node_attr @ W_in1 + b1) @ W_in2 + b2 (mirror to f16) ----
  gemmF(stream, node_attr, NODE_D, W_in1, b_in1, nullptr, t1, nullptr, NN, HD, NODE_D, 1);
  gemmF(stream, t1, HD, W_in2, b_in2, nullptr, ha, h16, NN, HD, HD, 0);

  // ---- message passing layers ----
  for (int l = 0; l < LAYERS; ++l) {
    const int KM = 2 * HD + RBF_DIM + EDGE_D; /* 292 */
    {
      int tiles = (EE / 16) * (HD / 32);
      msg_gemm<<<cdiv(tiles, 4), 128, 0, stream>>>(h16, rbf, edge_attr, srcI, dstI,
                                                   msg_W + (size_t)l * KM * HD,
                                                   msg_b + l * HD, m16);
    }
    agg_kernel<<<cdiv(NN * HD, 256), 256, 0, stream>>>(m16, agg);
    concat2<<<cdiv(NN * 256, 256), 256, 0, stream>>>(ha, agg, cat, NN, HD, HD);
    gemmF(stream, cat, 256, upd_W1 + (size_t)l * 256 * HD, upd_b1 + l * HD, nullptr,
          t1, nullptr, NN, HD, 256, 1);
    gemmF(stream, t1, HD, upd_W2 + (size_t)l * HD * HD, upd_b2 + l * HD, ha,
          intra, nullptr, NN, HD, HD, 0);
    gemmH(stream, m16, HD, x_W + (size_t)l * HD * HD, mx, EE, HD, HD);
    gemmH(stream, m16, HD, v_W + (size_t)l * HD * HD, mv, EE, HD, HD);
    vupd_kernel<<<cdiv(NN * HD, 256), 256, 0, stream>>>(va, mx, mv, dirv, srcI, vb);
    { float* tmp = va; va = vb; vb = tmp; }
    group_mean<<<8, HD, 0, stream>>>(intra, hbar);
    att_u<<<8, HD, 0, stream>>>(att_W + (size_t)l * HD * HD, hbar, uvec);
    att_gate<<<cdiv(NN, 256), 256, 0, stream>>>(intra, uvec, gate);
    gemmF(stream, intra, HD, act_W1 + (size_t)l * HD * HD, act_b1 + l * HD, nullptr,
          t1, nullptr, NN, HD, HD, 1);
    gemmF(stream, t1, HD, act_W2 + (size_t)l * HD * HD, act_b2 + l * HD, nullptr,
          t2, nullptr, NN, HD, HD, 0);
    h_update<<<cdiv(NN * HD, 256), 256, 0, stream>>>(intra, t2, gate, hb, h16);
    { float* tmp = ha; ha = hb; hb = tmp; }
  }

  // ---- gated equivariant block ----
  for (int c = 0; c < 3; ++c) {
    gemm_v3<<<cdiv((NN / 16) * 4, 4), 128, 0, stream>>>(va + c, geb_Wv1, v1 + c);
    gemm_v3<<<cdiv((NN / 16) * 4, 4), 128, 0, stream>>>(va + c, geb_Wv2, v2 + c);
  }
  vnorm_kernel<<<cdiv(NN * HD, 256), 256, 0, stream>>>(v2, t1);
  concat2<<<cdiv(NN * 256, 256), 256, 0, stream>>>(ha, t1, cat, NN, HD, HD);
  gemmF(stream, cat, 256, geb_W1, geb_b1, nullptr, t2, nullptr, NN, HD, 256, 1);
  gemmF(stream, t2, HD, geb_W2, geb_b2, nullptr, sbuf, nullptr, NN, 256, HD, 0);
  split_scale<<<cdiv(NN * HD, 256), 256, 0, stream>>>(sbuf, v1, hb, vb); /* hg=hb, vg=vb */

  // ---- keypoint generation ----
  cmean_kernel<<<1, 32, 0, stream>>>(Xca, cmean);
  tred_kernel<<<cdiv(8 * HD, 128), 128, 0, stream>>>(vb, hb, Tred);
  ykp_kernel<<<1, 96, 0, stream>>>(final_W, Tred, Yk);
  si_kernel<<<cdiv(NN, 128), 128, 0, stream>>>(vb, Xca, cmean, Si);
  gemmF(stream, Si, SI_LD, scale_W1, scale_b1, nullptr, t2, nullptr, NN, 64, 129, 1);
  denomrow_kernel<<<cdiv(NN, 256), 256, 0, stream>>>(t2, scale_W2, scale_b2, dnr);
  denom_kernel<<<8, 32, 0, stream>>>(dnr, denom);
  yfinal_kernel<<<1, 96, 0, stream>>>(Yk, denom, cmean, Yf);

  // ---- losses ----
  kabsch_kernel<<<BS, 32, 0, stream>>>(Yf, tkp, kp, rot, trans, RT, otdk);
  x1a_kernel<<<cdiv(BS * NP, 256), 256, 0, stream>>>(Xca, ori, RT, X1a, rmsdr);
  stab_kernel<<<cdiv(2 * BS * NP, 256), 256, 0, stream>>>(X1a, Xca, stabr);
  final_kernel<<<1, 32, 0, stream>>>(otdk, rmsdr, stabr, out);
}
